// ConvBNReLU1DFused_68968584839256
// MI455X (gfx1250) — matrix operand order repaired and checked by reading, not compiled, not run on hardware
//
#include <hip/hip_runtime.h>
#include <hip/hip_bf16.h>
#include <math.h>

// ---------------------------------------------------------------------------
// ConvBNReLU1DFused for MI455X (gfx1250, wave32, WMMA).
//   x: [B=32, Cin=512, N=2048] f32
//   W: [Cout=512, Cin=512] f32   -> po2 int8 fake-quant -> exact f16
//   b: [Cout=512] f32            -> po2 int8 fake-quant (kept f32)
//   y = relu(Wq @ x + bq)  -> global-max u8 fake-quant
// ---------------------------------------------------------------------------

#define B_SZ   32
#define CIN    512
#define COUT   512
#define NSP    2048   // spatial length

typedef __attribute__((ext_vector_type(16))) _Float16 v16h;
typedef __attribute__((ext_vector_type(8)))  _Float16 v8h;
typedef __attribute__((ext_vector_type(8)))  float    v8f;

union AFrag { v16h v; v8h h[2]; };

// ---------------------------------------------------------------------------
// Kernel 1: per-tensor po2 scales for W and b; quantize bias; init ymax.
// ---------------------------------------------------------------------------
__global__ __launch_bounds__(256)
void scales_kernel(const float* __restrict__ W, const float* __restrict__ b,
                   float* __restrict__ sW_out, float* __restrict__ bq_out,
                   unsigned int* __restrict__ ymax)
{
    __shared__ float red[256];
    const int tid = threadIdx.x;

    // max|W| over 262144 elements
    float mw = 0.f;
    for (int i = tid; i < COUT * CIN; i += 256) mw = fmaxf(mw, fabsf(W[i]));
    red[tid] = mw; __syncthreads();
    for (int s = 128; s > 0; s >>= 1) {
        if (tid < s) red[tid] = fmaxf(red[tid], red[tid + s]);
        __syncthreads();
    }
    const float sW = exp2f(ceilf(log2f(red[0] / 127.0f)));
    __syncthreads();

    // max|b| over 512 elements
    float mb = 0.f;
    for (int i = tid; i < COUT; i += 256) mb = fmaxf(mb, fabsf(b[i]));
    red[tid] = mb; __syncthreads();
    for (int s = 128; s > 0; s >>= 1) {
        if (tid < s) red[tid] = fmaxf(red[tid], red[tid + s]);
        __syncthreads();
    }
    const float sB = exp2f(ceilf(log2f(red[0] / 127.0f)));

    if (tid == 0) { *sW_out = sW; *ymax = 0u; }

    // fake-quant bias (internal scaling, round-half-even == rintf)
    const float invB = 1.0f / sB;
    for (int i = tid; i < COUT; i += 256) {
        float q = rintf(fminf(fmaxf(b[i] * invB, -127.f), 127.f));
        bq_out[i] = q * sB;
    }
}

// ---------------------------------------------------------------------------
// Kernel 2: quantize W -> exact f16 (values are int8 * 2^e, exact in f16).
// ---------------------------------------------------------------------------
__global__ __launch_bounds__(256)
void quantW_kernel(const float* __restrict__ W, const float* __restrict__ sWp,
                   _Float16* __restrict__ Wq)
{
    const float s = *sWp;
    const float inv = 1.0f / s;
    const int i = blockIdx.x * 256 + threadIdx.x;
    if (i < COUT * CIN) {
        float q = rintf(fminf(fmaxf(W[i] * inv, -127.f), 127.f));
        Wq[i] = (_Float16)(q * s);
    }
}

// ---------------------------------------------------------------------------
// Kernel 3: WMMA GEMM + bias + ReLU + global max.
// Block = 256 threads = 8 waves arranged 2(M) x 4(N).
// Block tile: 128 (M=Cout) x 128 (N=spatial).
// Wave tile:  64 x 32 -> 8 accumulators; each A fragment feeds 2 WMMAs,
// each B fragment feeds 4 WMMAs -> 8 v_wmma per K-step of 32.
// ---------------------------------------------------------------------------
__global__ __launch_bounds__(256)
void gemm_relu_kernel(const float* __restrict__ x,
                      const _Float16* __restrict__ Wq,
                      const float* __restrict__ bq,
                      float* __restrict__ y,
                      unsigned int* __restrict__ ymax)
{
    const int tid   = threadIdx.x;
    const int lane  = tid & 31;
    const int wave  = tid >> 5;
    const int col16 = lane & 15;   // column within 16x16 tile (B/N dim)
    const int khalf = lane >> 4;   // which K-half this lane holds

    const int bb = blockIdx.z;
    const int n0 = blockIdx.x * 128 + (wave & 3) * 32;   // spatial tile base
    const int m0 = blockIdx.y * 128 + (wave >> 2) * 64;  // Cout tile base

    const float* __restrict__ xb = x + (size_t)bb * CIN * NSP;

    v8f acc[4][2] = {};

    for (int kk = 0; kk < CIN; kk += 32) {
        // --- Two B fragments: 32(K) x 16(N) f16 each.
        // Lanes 0-15 hold K = kk+0..15 for column col16; lanes 16-31 K = kk+16..31.
        const float* xp = xb + (size_t)(kk + 16 * khalf) * NSP + n0 + col16;
        // Prefetch next K-slab of x (speculative; global_prefetch_b8).
        __builtin_prefetch(xp + (size_t)32 * NSP, 0, 1);
        v16h bfrag[2];
#pragma unroll
        for (int e = 0; e < 16; ++e) {
            bfrag[0][e] = (_Float16)xp[(size_t)e * NSP];
            bfrag[1][e] = (_Float16)xp[(size_t)e * NSP + 16];
        }

        // --- 4 A fragments (16x32 f16 each); each feeds both B fragments.
        // A layout (matches B): lanes 0-15 row M=lane: elems 0-15 -> K 0..15;
        // lanes 16-31 same rows, K 16..31.
#pragma unroll
        for (int i = 0; i < 4; ++i) {
            const _Float16* ap =
                Wq + (size_t)(m0 + 16 * i + col16) * CIN + kk + 16 * khalf;
            AFrag a;
            a.h[0] = *(const v8h*)(ap);        // K = kk + 16*khalf + 0..7
            a.h[1] = *(const v8h*)(ap + 8);    // K = kk + 16*khalf + 8..15
#pragma unroll
            for (int j = 0; j < 2; ++j) {
                acc[i][j] = __builtin_amdgcn_wmma_f32_16x16x32_f16(
                    false, a.v, false, bfrag[j], (short)0, acc[i][j],
                    false, false);
            }
        }
    }

    // --- Epilogue: bias + ReLU + store + running max.
    // D layout: lanes 0-15: M = v, N = lane; lanes 16-31: M = 8+v, N = lane-16.
    float lmax = 0.f;
    float* __restrict__ yb = y + (size_t)bb * COUT * NSP;
#pragma unroll
    for (int i = 0; i < 4; ++i) {
#pragma unroll
        for (int j = 0; j < 2; ++j) {
#pragma unroll
            for (int v = 0; v < 8; ++v) {
                const int row = m0 + 16 * i + 8 * khalf + v;
                float val = acc[i][j][v] + bq[row];
                val = fmaxf(val, 0.f);
                lmax = fmaxf(lmax, val);
                yb[(size_t)row * NSP + n0 + 16 * j + col16] = val;
            }
        }
    }

    __shared__ float red[256];
    red[tid] = lmax;
    __syncthreads();
    for (int s = 128; s > 0; s >>= 1) {
        if (tid < s) red[tid] = fmaxf(red[tid], red[tid + s]);
        __syncthreads();
    }
    // y >= 0, so uint ordering == float ordering.
    if (tid == 0) atomicMax(ymax, __float_as_uint(red[0]));
}

// ---------------------------------------------------------------------------
// Kernel 4: global-max unsigned-8-bit fake quant of output, in place.
// ---------------------------------------------------------------------------
__global__ __launch_bounds__(256)
void quant_out_kernel(float* __restrict__ y, const unsigned int* __restrict__ ymaxp,
                      long n4)
{
    const float m = __uint_as_float(*ymaxp);
    const float s = m * (1.0f / 255.0f);
    const float inv = (s > 0.f) ? (1.0f / s) : 0.f;
    float4* y4 = (float4*)y;
    long i = (long)blockIdx.x * blockDim.x + threadIdx.x;
    const long stride = (long)gridDim.x * blockDim.x;
    for (; i < n4; i += stride) {
        float4 v = y4[i];
        v.x = rintf(fminf(fmaxf(v.x * inv, 0.f), 255.f)) * s;
        v.y = rintf(fminf(fmaxf(v.y * inv, 0.f), 255.f)) * s;
        v.z = rintf(fminf(fmaxf(v.z * inv, 0.f), 255.f)) * s;
        v.w = rintf(fminf(fmaxf(v.w * inv, 0.f), 255.f)) * s;
        y4[i] = v;
    }
}

// ---------------------------------------------------------------------------
// Launch
// ---------------------------------------------------------------------------
extern "C" void kernel_launch(void* const* d_in, const int* in_sizes, int n_in,
                              void* d_out, int out_size, void* d_ws, size_t ws_size,
                              hipStream_t stream)
{
    const float* x = (const float*)d_in[0];
    const float* W = (const float*)d_in[1];
    const float* b = (const float*)d_in[2];
    float* out = (float*)d_out;

    // Workspace layout:
    //   [0]    float sW
    //   [8]    uint  ymax
    //   [16]   float bq[512]
    //   [4096] _Float16 Wq[512*512]   (512 KB; hot in L2)
    char* ws = (char*)d_ws;
    float*        sW   = (float*)(ws + 0);
    unsigned int* ymax = (unsigned int*)(ws + 8);
    float*        bq   = (float*)(ws + 16);
    _Float16*     Wq   = (_Float16*)(ws + 4096);

    scales_kernel<<<1, 256, 0, stream>>>(W, b, sW, bq, ymax);
    quantW_kernel<<<(COUT * CIN) / 256, 256, 0, stream>>>(W, sW, Wq);

    dim3 grid(NSP / 128, COUT / 128, B_SZ);   // (16, 4, 32) blocks
    gemm_relu_kernel<<<grid, 256, 0, stream>>>(x, Wq, bq, out, ymax);

    const long n4 = (long)B_SZ * COUT * NSP / 4;
    quant_out_kernel<<<2048, 256, 0, stream>>>(out, ymax, n4);
}
